// SemanticRQVAE_3968549782349
// MI455X (gfx1250) — matrix operand order repaired and checked by reading, not compiled, not run on hardware
//
#include <hip/hip_runtime.h>
#include <hip/hip_bf16.h>
#include <stddef.h>

typedef __bf16 bf16_t;
typedef __attribute__((ext_vector_type(16))) __bf16 v16bf;
typedef __attribute__((ext_vector_type(8)))  __bf16 v8bf;
typedef __attribute__((ext_vector_type(8)))  float  v8f;

#define BATCH   65536
#define EMBED   384
#define HID     512
#define CBK     256
#define NQ      4
#define ROWS    16
#define THREADS 256
#define NBLK    (BATCH / ROWS)   // 4096
#define PERT    ((ROWS * HID) / THREADS)  // 32 elements owned per thread

// ---- d_ws layout (bytes) ----
#define OFF_W1T  0                                   // [512][384] bf16 (enc_w1^T)
#define OFF_W2T  (OFF_W1T + 512 * 384 * 2)           // [512][512] bf16 (enc_w2^T)
#define OFF_D1T  (OFF_W2T + 512 * 512 * 2)           // [512][512] bf16 (dec_w1^T)
#define OFF_D2T  (OFF_D1T + 512 * 512 * 2)           // [384][512] bf16 (dec_w2^T)
#define OFF_CB   (OFF_D2T + 384 * 512 * 2)           // [4][256][512] bf16 codebooks
#define OFF_CN   (OFF_CB + 4 * 256 * 512 * 2)        // [4][256] f32 ||c||^2
#define OFF_PART (OFF_CN + 4 * 256 * 4)              // [NBLK][2] f32 partial losses

// ======================= prep kernels =======================

// W: [K][N] f32  ->  Wt: [N][K] bf16
__global__ void k_transpose_bf16(const float* __restrict__ W, bf16_t* __restrict__ Wt,
                                 int K, int N) {
  int i = blockIdx.x * 256 + threadIdx.x;
  if (i >= K * N) return;
  int n = i / K, k = i % K;
  Wt[i] = (bf16_t)W[k * N + n];
}

__global__ void k_cvt_cb(const float* __restrict__ cb, bf16_t* __restrict__ cbb, int total) {
  int i = blockIdx.x * 256 + threadIdx.x;
  if (i < total) cbb[i] = (bf16_t)cb[i];
}

__global__ void k_cnorm(const float* __restrict__ cb, float* __restrict__ cn) {
  int i = blockIdx.x * 256 + threadIdx.x;   // one thread per (q, code)
  if (i >= NQ * CBK) return;
  const float* p = cb + (size_t)i * HID;
  float s = 0.f;
  for (int k = 0; k < HID; ++k) { float v = p[k]; s += v * v; }
  cn[i] = s;
}

// ======================= WMMA fragment helpers =======================

// A: 16x32 bf16 tile, row-major in LDS with leading dim lda.
// lane L: m = L%16, g = L/16; elems 0..7 -> K = kbase + g*8 + e, elems 8..15 -> +16.
__device__ __forceinline__ v16bf load_a_frag(const bf16_t* __restrict__ sA, int lda,
                                             int kbase, int lane) {
  int m = lane & 15, g = lane >> 4;
  const bf16_t* p = sA + m * lda + kbase + g * 8;
  v8bf lo = *(const v8bf*)p;
  v8bf hi = *(const v8bf*)(p + 16);
  v16bf f;
#pragma unroll
  for (int i = 0; i < 8; ++i) { f[i] = lo[i]; f[i + 8] = hi[i]; }
  return f;
}

// B: 32x16 bf16 tile from pre-transposed weights Bt[N][K] (row length ldk).
// lane L: n = ncol0 + L%16, g = L/16; elems 0..7 -> K = kbase + g*8 + e, elems 8..15 -> +16.
__device__ __forceinline__ v16bf load_b_frag(const bf16_t* __restrict__ Bt, int ldk,
                                             int n, int kbase, int lane) {
  int g = lane >> 4;
  const bf16_t* p = Bt + (size_t)n * ldk + kbase + g * 8;
  v8bf lo = *(const v8bf*)p;
  v8bf hi = *(const v8bf*)(p + 16);
  v16bf f;
#pragma unroll
  for (int i = 0; i < 8; ++i) { f[i] = lo[i]; f[i + 8] = hi[i]; }
  return f;
}

// C = act(A(16xK) @ Bt^T + bias) into LDS f32 [16][ldc]; waves stride N-tiles.
template <bool RELU>
__device__ __forceinline__ void block_gemm(const bf16_t* __restrict__ sA, int lda,
                                           const bf16_t* __restrict__ Bt, int K, int N,
                                           const float* __restrict__ bias,
                                           float* __restrict__ sC, int ldc, int tid) {
  const int w = tid >> 5, lane = tid & 31;
  const int g = lane >> 4, nl = lane & 15;
  for (int nt = w; nt < (N >> 4); nt += 8) {
    v8f acc = {};
    for (int k = 0; k < K; k += 32) {
      v16bf a = load_a_frag(sA, lda, k, lane);
      v16bf b = load_b_frag(Bt, K, nt * 16 + nl, k, lane);
      acc = __builtin_amdgcn_wmma_f32_16x16x32_bf16(false, a, false, b, (short)0, acc,
                                                    false, false);
    }
    float bv = bias[nt * 16 + nl];
#pragma unroll
    for (int r = 0; r < 8; ++r) {
      float v = acc[r] + bv;
      if (RELU) v = fmaxf(v, 0.0f);
      sC[(g * 8 + r) * ldc + nt * 16 + nl] = v;
    }
  }
}

// ======================= fused RQ-VAE kernel =======================

__global__ __launch_bounds__(THREADS) void k_fused(
    const float* __restrict__ x,
    const float* __restrict__ eb1, const float* __restrict__ eb2,
    const float* __restrict__ cb,                    // f32 codebooks [4][256][512]
    const float* __restrict__ db1, const float* __restrict__ db2,
    const bf16_t* __restrict__ w1t, const bf16_t* __restrict__ w2t,
    const bf16_t* __restrict__ d1t, const bf16_t* __restrict__ d2t,
    const bf16_t* __restrict__ cbb, const float* __restrict__ cnorm,
    float* __restrict__ out_recon, float* __restrict__ out_idx,
    float* __restrict__ partials) {
  __shared__ __attribute__((aligned(16))) bf16_t sAct[ROWS * HID];  // 16 KB: A operand
  __shared__ __attribute__((aligned(16))) float  sC[ROWS * HID];    // 32 KB: f32 stage/residual
  __shared__ float sCandV[ROWS][8];
  __shared__ int   sCandI[ROWS][8];
  __shared__ int   sIdx[ROWS];
  __shared__ float sRed[THREADS];

  const int tid = threadIdx.x;
  const int w = tid >> 5, lane = tid & 31, g = lane >> 4, nl = lane & 15;
  const int row0 = blockIdx.x * ROWS;

  // ---- load x tile -> bf16 LDS ----
#pragma unroll
  for (int j = 0; j < (ROWS * EMBED) / THREADS; ++j) {   // 24
    int e = tid + THREADS * j;
    int r = e / EMBED, c = e % EMBED;
    sAct[r * HID + c] = (bf16_t)x[(size_t)(row0 + r) * EMBED + c];
  }
  __syncthreads();

  // ---- encoder layer 1: relu(x @ W1 + b1) ----
  block_gemm<true>(sAct, HID, w1t, EMBED, HID, eb1, sC, HID, tid);
  __syncthreads();
#pragma unroll
  for (int j = 0; j < PERT; ++j) { int e = tid + THREADS * j; sAct[e] = (bf16_t)sC[e]; }
  __syncthreads();

  // ---- encoder layer 2: z = h @ W2 + b2 (f32 residual in sC) ----
  block_gemm<false>(sAct, HID, w2t, HID, HID, eb2, sC, HID, tid);
  __syncthreads();
  float quant[PERT];
#pragma unroll
  for (int j = 0; j < PERT; ++j) {
    int e = tid + THREADS * j;
    sAct[e] = (bf16_t)sC[e];
    quant[j] = 0.0f;
  }
  __syncthreads();

  float commit_acc = 0.0f;

  // ---- residual VQ, 4 stages ----
  for (int q = 0; q < NQ; ++q) {
    const bf16_t* cbq = cbb + (size_t)q * CBK * HID;
    const float*  cnq = cnorm + q * CBK;

    float bestV[8]; int bestI[8];
#pragma unroll
    for (int r = 0; r < 8; ++r) { bestV[r] = -3.0e38f; bestI[r] = 0; }

    for (int nt = w; nt < (CBK >> 4); nt += 8) {   // nt = w, w+8
      v8f acc = {};
      for (int k = 0; k < HID; k += 32) {
        v16bf a = load_a_frag(sAct, HID, k, lane);
        v16bf b = load_b_frag(cbq, HID, nt * 16 + nl, k, lane);
        acc = __builtin_amdgcn_wmma_f32_16x16x32_bf16(false, a, false, b, (short)0, acc,
                                                      false, false);
      }
#pragma unroll
      for (int r = 0; r < 8; ++r) {
        float v = 2.0f * acc[r] - cnq[nt * 16 + nl];
        int idx = nt * 16 + nl;
        // butterfly max across the 16-lane column group (first-max tie-break)
#pragma unroll
        for (int m = 1; m < 16; m <<= 1) {
          float ov = __shfl_xor(v, m, 32);
          int   oi = __shfl_xor(idx, m, 32);
          if (ov > v || (ov == v && oi < idx)) { v = ov; idx = oi; }
        }
        if (v > bestV[r] || (v == bestV[r] && idx < bestI[r])) { bestV[r] = v; bestI[r] = idx; }
      }
    }
    if (nl == 0) {
#pragma unroll
      for (int r = 0; r < 8; ++r) { sCandV[g * 8 + r][w] = bestV[r]; sCandI[g * 8 + r][w] = bestI[r]; }
    }
    __syncthreads();
    if (tid < ROWS) {
      float bv = sCandV[tid][0]; int bi = sCandI[tid][0];
#pragma unroll
      for (int c = 1; c < 8; ++c) {
        float v = sCandV[tid][c]; int i2 = sCandI[tid][c];
        if (v > bv || (v == bv && i2 < bi)) { bv = v; bi = i2; }
      }
      sIdx[tid] = bi;
      out_idx[(size_t)(row0 + tid) * NQ + q] = (float)bi;
    }
    __syncthreads();

    // commit loss + residual update + quantized accumulation (f32 codebook gather)
#pragma unroll
    for (int j = 0; j < PERT; ++j) {
      int e = tid + THREADS * j;
      int r = e >> 9, c = e & (HID - 1);
      float qv = cb[((size_t)q * CBK + sIdx[r]) * HID + c];
      float res = sC[e];
      float d = res - qv;
      commit_acc += d * d;
      res -= qv;
      sC[e] = res;
      sAct[e] = (bf16_t)res;
      quant[j] += qv;
    }
    __syncthreads();
  }

  // ---- decoder input = quantized ----
#pragma unroll
  for (int j = 0; j < PERT; ++j) { int e = tid + THREADS * j; sAct[e] = (bf16_t)quant[j]; }
  __syncthreads();

  block_gemm<true>(sAct, HID, d1t, HID, HID, db1, sC, HID, tid);
  __syncthreads();
#pragma unroll
  for (int j = 0; j < PERT; ++j) { int e = tid + THREADS * j; sAct[e] = (bf16_t)sC[e]; }
  __syncthreads();

  // ---- decoder layer 2 -> recon to global + recon loss partial ----
  float recon_acc = 0.0f;
  for (int nt = w; nt < (EMBED >> 4); nt += 8) {   // 24 tiles
    v8f acc = {};
    for (int k = 0; k < HID; k += 32) {
      v16bf a = load_a_frag(sAct, HID, k, lane);
      v16bf b = load_b_frag(d2t, HID, nt * 16 + nl, k, lane);
      acc = __builtin_amdgcn_wmma_f32_16x16x32_bf16(false, a, false, b, (short)0, acc,
                                                    false, false);
    }
    float bv = db2[nt * 16 + nl];
#pragma unroll
    for (int r = 0; r < 8; ++r) {
      int m = g * 8 + r, n = nt * 16 + nl;
      float v = acc[r] + bv;
      size_t o = (size_t)(row0 + m) * EMBED + n;
      out_recon[o] = v;
      float d = v - x[o];
      recon_acc += d * d;
    }
  }

  // ---- deterministic block reductions -> partials ----
  sRed[tid] = recon_acc;
  __syncthreads();
  for (int s = THREADS / 2; s > 0; s >>= 1) {
    if (tid < s) sRed[tid] += sRed[tid + s];
    __syncthreads();
  }
  if (tid == 0) partials[blockIdx.x * 2 + 0] = sRed[0];
  __syncthreads();
  sRed[tid] = commit_acc;
  __syncthreads();
  for (int s = THREADS / 2; s > 0; s >>= 1) {
    if (tid < s) sRed[tid] += sRed[tid + s];
    __syncthreads();
  }
  if (tid == 0) partials[blockIdx.x * 2 + 1] = sRed[0];
}

// ======================= final loss reduction =======================

__global__ __launch_bounds__(256) void k_reduce(const float* __restrict__ partials,
                                                float* __restrict__ out_scalars) {
  __shared__ float sA[256], sB[256];
  int t = threadIdx.x;
  float ra = 0.f, ca = 0.f;
  for (int b = t; b < NBLK; b += 256) { ra += partials[b * 2]; ca += partials[b * 2 + 1]; }
  sA[t] = ra; sB[t] = ca;
  __syncthreads();
  for (int s = 128; s > 0; s >>= 1) {
    if (t < s) { sA[t] += sA[t + s]; sB[t] += sB[t + s]; }
    __syncthreads();
  }
  if (t == 0) {
    out_scalars[0] = sA[0] / ((float)BATCH * (float)EMBED);              // recon_loss
    out_scalars[1] = 0.25f * sB[0] / ((float)BATCH * (float)HID);       // commit_loss
  }
}

// ======================= host entry =======================

extern "C" void kernel_launch(void* const* d_in, const int* in_sizes, int n_in,
                              void* d_out, int out_size, void* d_ws, size_t ws_size,
                              hipStream_t stream) {
  const float* x   = (const float*)d_in[0];
  const float* ew1 = (const float*)d_in[1];
  const float* eb1 = (const float*)d_in[2];
  const float* ew2 = (const float*)d_in[3];
  const float* eb2 = (const float*)d_in[4];
  const float* cb  = (const float*)d_in[5];
  const float* dw1 = (const float*)d_in[6];
  const float* db1 = (const float*)d_in[7];
  const float* dw2 = (const float*)d_in[8];
  const float* db2 = (const float*)d_in[9];

  char* ws = (char*)d_ws;
  bf16_t* w1t = (bf16_t*)(ws + OFF_W1T);
  bf16_t* w2t = (bf16_t*)(ws + OFF_W2T);
  bf16_t* d1t = (bf16_t*)(ws + OFF_D1T);
  bf16_t* d2t = (bf16_t*)(ws + OFF_D2T);
  bf16_t* cbb = (bf16_t*)(ws + OFF_CB);
  float*  cn  = (float*)(ws + OFF_CN);
  float*  prt = (float*)(ws + OFF_PART);

  float* out_recon = (float*)d_out;
  float* out_idx   = out_recon + (size_t)BATCH * EMBED;
  float* out_scal  = out_idx + (size_t)BATCH * NQ;

  // weight prep: f32 [K][N] -> bf16 [N][K]; codebooks -> bf16; ||c||^2
  k_transpose_bf16<<<(EMBED * HID + 255) / 256, 256, 0, stream>>>(ew1, w1t, EMBED, HID);
  k_transpose_bf16<<<(HID * HID + 255) / 256, 256, 0, stream>>>(ew2, w2t, HID, HID);
  k_transpose_bf16<<<(HID * HID + 255) / 256, 256, 0, stream>>>(dw1, d1t, HID, HID);
  k_transpose_bf16<<<(HID * EMBED + 255) / 256, 256, 0, stream>>>(dw2, d2t, HID, EMBED);
  k_cvt_cb<<<(NQ * CBK * HID + 255) / 256, 256, 0, stream>>>(cb, cbb, NQ * CBK * HID);
  k_cnorm<<<(NQ * CBK + 255) / 256, 256, 0, stream>>>(cb, cn);

  // fused encoder + residual VQ + decoder + loss partials
  k_fused<<<NBLK, THREADS, 0, stream>>>(x, eb1, eb2, cb, db1, db2,
                                        w1t, w2t, d1t, d2t, cbb, cn,
                                        out_recon, out_idx, prt);

  // deterministic final reduction of the two scalar losses
  k_reduce<<<1, 256, 0, stream>>>(prt, out_scal);
}
